// BiAttentionLayer_68693706932834
// MI455X (gfx1250) — matrix operand order repaired and checked by reading, not compiled, not run on hardware
//
#include <hip/hip_runtime.h>

// ---- MI455X (gfx1250) flash-style BiDAF attention ----------------------------
// S[i,j] = chat_i . q_j + qb[j]   (row-constant c.w_c term cancels in softmax)
// flash softmax => U_A rows and b[i] = 1/l_i for free; h = sum_i b[i]*c[i].
// Compute-bound: 2 * T^2 * D * 2 = 34.4 GFLOP -> f16 WMMA (16x16x32), f32 accum.
// Tile fills use gfx1250 async global->LDS DMA (ASYNCcnt) when available.

typedef _Float16 v16h __attribute__((ext_vector_type(16)));
typedef _Float16 v8h  __attribute__((ext_vector_type(8)));
typedef float    v8f  __attribute__((ext_vector_type(8)));
typedef float    v4f  __attribute__((ext_vector_type(4)));

constexpr int kT  = 8192;
constexpr int kD  = 128;
constexpr int kBM = 64;    // rows per workgroup (4 waves x 16 rows)
constexpr int kBN = 128;   // key columns per iteration
constexpr int kNW = 4;     // waves per workgroup

// gfx1250 async global->LDS path (ASYNCcnt), guarded: falls back to reg copies.
#if defined(__has_builtin)
#if __has_builtin(__builtin_amdgcn_global_load_async_to_lds_b128) && \
    __has_builtin(__builtin_amdgcn_s_wait_asynccnt)
#define USE_ASYNC_LDS 1
#endif
#endif
#ifndef USE_ASYNC_LDS
#define USE_ASYNC_LDS 0
#endif

#if USE_ASYNC_LDS
// builtin prototype (from hipcc diagnostic): (v4i AS1*, v4i AS3*, imm i32, imm i32)
typedef int v4i __attribute__((vector_size(4 * sizeof(int))));
typedef __attribute__((address_space(1))) v4i glb_v4i;
typedef __attribute__((address_space(3))) v4i lds_v4i;

__device__ __forceinline__ void async_copy16(const void* gsrc, void* ldst)
{
    __builtin_amdgcn_global_load_async_to_lds_b128(
        (glb_v4i*)(v4i*)gsrc, (lds_v4i*)(v4i*)ldst, 0, 0);
}
#endif

// ---- pass 0: convert to f16 (chat = c*w_m, q row-major, q transposed), qb ----
__global__ __launch_bounds__(128)
void prep_kernel(const float* __restrict__ x, const float* __restrict__ w,
                 _Float16* __restrict__ chat, _Float16* __restrict__ q16,
                 _Float16* __restrict__ qtr, float* __restrict__ qb)
{
    __shared__ float red[128];
    const int r = blockIdx.x;
    const int d = threadIdx.x;
    const float c = x[(size_t)r * kD + d];
    const float q = x[(size_t)kT * kD + (size_t)r * kD + d];
    chat[(size_t)r * kD + d] = (_Float16)(c * w[2 * kD + d]);
    q16 [(size_t)r * kD + d] = (_Float16)q;
    qtr [(size_t)d * kT + r] = (_Float16)q;            // transposed copy for A@V B-frags
    red[d] = q * w[kD + d];
    __syncthreads();
    #pragma unroll
    for (int s = 64; s > 0; s >>= 1) {
        if (d < s) red[d] += red[d + s];
        __syncthreads();
    }
    if (d == 0) qb[r] = red[0];
}

// ---- pass 1: flash attention -------------------------------------------------
__global__ __launch_bounds__(128)
void attn_pass1(const _Float16* __restrict__ chat, const _Float16* __restrict__ qrow,
                const _Float16* __restrict__ qtr,  const float* __restrict__ qb,
                float* __restrict__ ua, float* __restrict__ binv)
{
    __shared__ _Float16 lds_q [kBN * kD];          // question tile, (j,d) row-major, 32KB
    __shared__ _Float16 lds_qt[kD * kBN];          // question tile, (d,j) col-major, 32KB
    __shared__ float    lds_qb[kBN];
    __shared__ _Float16 lds_p [kNW * 16 * kBN];    // per-wave P strip (f16), 16KB

    const int tid  = threadIdx.x;
    const int wave = tid >> 5;
    const int lane = tid & 31;
    const int l16  = lane & 15;
    const int hi   = lane >> 4;                    // half-wave select

    const int i0 = blockIdx.x * kBM + wave * 16;   // this wave's 16-row strip

    // A-fragments of chat (16 x 128 = 4 frags of 16x32), contiguous K order:
    // lanes 0-15: halves 0-15 -> K=0..15 ; lanes 16-31: K=16..31.
    v16h afrag[4];
    {
        const int row = i0 + l16;
        #pragma unroll
        for (int k = 0; k < 4; ++k) {
            const int d0 = 32 * k + hi * 16;
            union { v16h v; v8h h[2]; } u;
            u.h[0] = *(const v8h*)(chat + (size_t)row * kD + d0);
            u.h[1] = *(const v8h*)(chat + (size_t)row * kD + d0 + 8);
            afrag[k] = u.v;
        }
    }

    float m[8], l[8];
    v8f   o[8];
    #pragma unroll
    for (int r = 0; r < 8; ++r) { m[r] = -__builtin_inff(); l[r] = 0.0f; o[r] = (v8f){}; }

    const float LOG2E = 1.4426950408889634f;

    for (int jb = 0; jb < kT / kBN; ++jb) {
        const int j0 = jb * kBN;
        __syncthreads();   // previous iteration done reading lds_q/lds_qt
        {   // cooperative tile fill: thread t owns row j=t (row tile) and col d=t (T tile)
            const _Float16* src  = qrow + (size_t)(j0 + tid) * kD;
            _Float16*       dst  = lds_q + tid * kD;
            const _Float16* srct = qtr + (size_t)tid * kT + j0;
            _Float16*       dstt = lds_qt + tid * kBN;
#if USE_ASYNC_LDS
            // direct cache->LDS DMA, 16B per lane per op, tracked by ASYNCcnt
            #pragma unroll
            for (int c = 0; c < kD; c += 8)
                async_copy16(src + c, dst + c);
            #pragma unroll
            for (int c = 0; c < kBN; c += 8)
                async_copy16(srct + c, dstt + c);
            lds_qb[tid] = qb[j0 + tid];
            __builtin_amdgcn_s_wait_asynccnt(0);
#else
            #pragma unroll
            for (int c = 0; c < kD; c += 16)
                *(v16h*)(dst + c) = *(const v16h*)(src + c);
            #pragma unroll
            for (int c = 0; c < kBN; c += 16)
                *(v16h*)(dstt + c) = *(const v16h*)(srct + c);
            lds_qb[tid] = qb[j0 + tid];
#endif
        }
        __syncthreads();

        // ---- S strip = chat @ q^T + qb : 8 tiles of 16x16, K=128 ----
        v8f s[8];
        #pragma unroll
        for (int nt = 0; nt < 8; ++nt) {
            v8f acc = (v8f){};
            const int j = nt * 16 + l16;           // B-frag: N = j, K contiguous
            #pragma unroll
            for (int k = 0; k < 4; ++k) {
                v16h b = *(const v16h*)(lds_q + j * kD + 32 * k + hi * 16);
                acc = __builtin_amdgcn_wmma_f32_16x16x32_f16(
                          false, afrag[k], false, b, (short)0, acc, false, false);
            }
            const float qbv = lds_qb[nt * 16 + l16];
            #pragma unroll
            for (int r = 0; r < 8; ++r) acc[r] = acc[r] + qbv;
            s[nt] = acc;
        }

        // ---- online softmax (rows live across 16 lanes of each half-wave) ----
        float scale[8];
        #pragma unroll
        for (int r = 0; r < 8; ++r) {
            float tm = s[0][r];
            #pragma unroll
            for (int nt = 1; nt < 8; ++nt) tm = fmaxf(tm, s[nt][r]);
            tm = fmaxf(tm, __shfl_xor(tm, 1));
            tm = fmaxf(tm, __shfl_xor(tm, 2));
            tm = fmaxf(tm, __shfl_xor(tm, 4));
            tm = fmaxf(tm, __shfl_xor(tm, 8));
            const float mnew = fmaxf(m[r], tm);
            scale[r] = __builtin_amdgcn_exp2f((m[r] - mnew) * LOG2E);
            m[r] = mnew;
        }

        _Float16* pbase = lds_p + wave * 16 * kBN;
        float rs[8];
        #pragma unroll
        for (int r = 0; r < 8; ++r) rs[r] = 0.0f;
        #pragma unroll
        for (int nt = 0; nt < 8; ++nt) {
            #pragma unroll
            for (int r = 0; r < 8; ++r) {
                const float p = __builtin_amdgcn_exp2f((s[nt][r] - m[r]) * LOG2E);
                rs[r] += p;
                pbase[(r + hi * 8) * kBN + nt * 16 + l16] = (_Float16)p;
            }
        }
        #pragma unroll
        for (int r = 0; r < 8; ++r) {
            float t = rs[r];
            t += __shfl_xor(t, 1); t += __shfl_xor(t, 2);
            t += __shfl_xor(t, 4); t += __shfl_xor(t, 8);
            l[r] = l[r] * scale[r] + t;
            #pragma unroll
            for (int nt = 0; nt < 8; ++nt) o[nt][r] = o[nt][r] * scale[r];
        }

        // ---- A-fragments of P from LDS (same contiguous layout) ----
        v16h pf[4];
        #pragma unroll
        for (int k = 0; k < 4; ++k) {
            const int d0 = 32 * k + hi * 16;
            union { v16h v; v8h h[2]; } u;
            u.h[0] = *(const v8h*)(pbase + l16 * kBN + d0);
            u.h[1] = *(const v8h*)(pbase + l16 * kBN + d0 + 8);
            pf[k] = u.v;
        }

        // ---- O += P @ Qtile (B-frags from transposed tile: K=j contiguous) ----
        #pragma unroll
        for (int nt = 0; nt < 8; ++nt) {
            v8f acc = o[nt];
            const int d = nt * 16 + l16;           // B-frag: N = d
            #pragma unroll
            for (int k = 0; k < 4; ++k) {
                v16h b = *(const v16h*)(lds_qt + d * kBN + 32 * k + hi * 16);
                acc = __builtin_amdgcn_wmma_f32_16x16x32_f16(
                          false, pf[k], false, b, (short)0, acc, false, false);
            }
            o[nt] = acc;
        }
    }

    // ---- epilogue: U_A = O / l ; binv = 1/l (== max_j A[i,j]) ----
    float linv[8];
    #pragma unroll
    for (int r = 0; r < 8; ++r) linv[r] = 1.0f / l[r];
    #pragma unroll
    for (int nt = 0; nt < 8; ++nt)
        #pragma unroll
        for (int r = 0; r < 8; ++r)
            ua[(size_t)(i0 + r + hi * 8) * kD + nt * 16 + l16] = o[nt][r] * linv[r];
    if (l16 == 0) {
        #pragma unroll
        for (int r = 0; r < 8; ++r)
            binv[i0 + r + hi * 8] = linv[r];
    }
}

// ---- pass 2: h = sum_i binv[i] * context[i] (deterministic 2-level) ----------
__global__ __launch_bounds__(128)
void hpartial_kernel(const float* __restrict__ x, const float* __restrict__ binv,
                     float* __restrict__ hp)
{
    const int b = blockIdx.x, d = threadIdx.x;
    float s = 0.0f;
    for (int i = 0; i < 64; ++i) {
        const int row = b * 64 + i;
        s += binv[row] * x[(size_t)row * kD + d];
    }
    hp[b * kD + d] = s;
}

__global__ __launch_bounds__(128)
void hreduce_kernel(const float* __restrict__ hp, float* __restrict__ h)
{
    const int d = threadIdx.x;
    float s = 0.0f;
    for (int b = 0; b < 128; ++b) s += hp[b * kD + d];
    h[d] = s;
}

// ---- pass 3: G = [c, U_A, c*U_A, c*h] ---------------------------------------
__global__ __launch_bounds__(256)
void assemble_kernel(const float* __restrict__ x, const float* __restrict__ ua,
                     const float* __restrict__ h, float* __restrict__ out)
{
    const int gid = blockIdx.x * 256 + threadIdx.x;   // row*32 + d4
    const int row = gid >> 5;
    const int d   = (gid & 31) * 4;
    const v4f c  = *(const v4f*)(x  + (size_t)row * kD + d);
    const v4f u  = *(const v4f*)(ua + (size_t)row * kD + d);
    const v4f hv = *(const v4f*)(h + d);
    float* o = out + (size_t)row * (4 * kD);
    *(v4f*)(o + d)            = c;
    *(v4f*)(o + kD + d)       = u;
    *(v4f*)(o + 2 * kD + d)   = c * u;
    *(v4f*)(o + 3 * kD + d)   = c * hv;
}

extern "C" void kernel_launch(void* const* d_in, const int* in_sizes, int n_in,
                              void* d_out, int out_size, void* d_ws, size_t ws_size,
                              hipStream_t stream)
{
    const float* x = (const float*)d_in[0];      // (2,1,T,D) f32
    const float* w = (const float*)d_in[1];      // (3*D,)    f32

    char* ws = (char*)d_ws;
    const size_t MB = 1024u * 1024u;
    _Float16* chat = (_Float16*)(ws);                       // 2 MB
    _Float16* q16  = (_Float16*)(ws + 2 * MB);              // 2 MB
    _Float16* qtr  = (_Float16*)(ws + 4 * MB);              // 2 MB
    float*    ua   = (float*)   (ws + 6 * MB);              // 4 MB
    float*    qb   = (float*)   (ws + 10 * MB);             // 32 KB
    float*    binv = (float*)   (ws + 10 * MB + 32 * 1024); // 32 KB
    float*    hp   = (float*)   (ws + 10 * MB + 64 * 1024); // 64 KB
    float*    h    = (float*)   (ws + 10 * MB + 128 * 1024);// 512 B

    prep_kernel     <<<kT, 128, 0, stream>>>(x, w, chat, q16, qtr, qb);
    attn_pass1      <<<kT / kBM, 128, 0, stream>>>(chat, q16, qtr, qb, ua, binv);
    hpartial_kernel <<<128, 128, 0, stream>>>(x, binv, hp);
    hreduce_kernel  <<<1, 128, 0, stream>>>(hp, h);
    assemble_kernel <<<(kT * 32) / 256, 256, 0, stream>>>(x, ua, h, (float*)d_out);
}
